// Gpt2Attention_27041114096448
// MI455X (gfx1250) — compile-verified, matrix operand order checked by reading, not run on hardware
//
#include <hip/hip_runtime.h>
#include <cstdint>
#include <cstddef>

// ---------------------------------------------------------------------------
// GPT-2 attention block for gfx1250 (MI455X).
// bf16 WMMA + fp32 accumulate; bf16 intermediates; async global->LDS staging.
// Pipeline: cvt(x,w_qkv,w_proj) -> QKV GEMM -> flash attention -> proj GEMM.
// ---------------------------------------------------------------------------

typedef __attribute__((ext_vector_type(16))) __bf16         v16bf;
typedef __attribute__((ext_vector_type(16))) unsigned short v16u;
typedef __attribute__((ext_vector_type(8)))  float          v8f;

static constexpr int kB  = 2;
static constexpr int kS  = 2048;
static constexpr int kD  = 1024;
static constexpr int kH  = 16;
static constexpr int kHS = 64;
static constexpr int kM    = kB * kS;        // 4096 token rows
static constexpr int kNqkv = 3 * kH * kHS;   // 3072

__device__ __forceinline__ unsigned short f32_to_bf16(float f) {
  union { float f; unsigned int u; } cv; cv.f = f;
  unsigned int u = cv.u;
  u += 0x7FFFu + ((u >> 16) & 1u);           // round-to-nearest-even
  return (unsigned short)(u >> 16);
}

// Async DMA of one 16-byte chunk global -> LDS (raw copy, no VGPR data path).
// Tracked by ASYNCcnt; pair with async_wait0() before the consuming barrier.
__device__ __forceinline__ void async_copy16(unsigned short* lds_dst,
                                             const unsigned short* g_src) {
  unsigned lds = (unsigned)(uintptr_t)lds_dst;           // low 32b = LDS offset
  unsigned long long ga = (unsigned long long)(uintptr_t)g_src;
  asm volatile("global_load_async_to_lds_b128 %0, %1, off"
               :: "v"(lds), "v"(ga) : "memory");
}
__device__ __forceinline__ void async_wait0() {
  asm volatile("s_wait_asynccnt 0x0" ::: "memory");
}

// Load a 16x32 bf16 WMMA operand fragment from LDS.
// A-matrix: row = M (lane&15), cols = K.  B-matrix stored transposed in LDS so
// row = N.  Per-lane K pattern per CDNA5 ISA 7.12.2 (hi = lane>=16):
//   VGPR i<4 : K = 2i + 8*hi ; VGPR i>=4: K = 16 + 2(i-4) + 8*hi
__device__ __forceinline__ v16bf load_frag(const unsigned short* base, int lane,
                                           int stride /*in shorts*/) {
  const unsigned short* row = base + (lane & 15) * stride;
  const int hi8 = ((lane >> 4) & 1) * 8;
  v16u u;
#pragma unroll
  for (int i = 0; i < 8; ++i) {
    const int kb = ((i < 4) ? 2 * i : 16 + 2 * (i - 4)) + hi8;
    u[2 * i]     = row[kb];
    u[2 * i + 1] = row[kb + 1];
  }
  return __builtin_bit_cast(v16bf, u);
}

__device__ __forceinline__ v8f wmma_bf16(v16bf a, v16bf b, v8f c) {
  return __builtin_amdgcn_wmma_f32_16x16x32_bf16(false, a, false, b,
                                                 (short)0, c, false, false);
}

// ---------------------------------------------------------------------------
// Elementwise fp32 -> bf16 (one-time conversion of inputs/weights).
// ---------------------------------------------------------------------------
__global__ __launch_bounds__(256) void cvt_f32_bf16(const float* __restrict__ src,
                                                    unsigned short* __restrict__ dst,
                                                    int n) {
  int i = (blockIdx.x * 256 + threadIdx.x) * 4;
  if (i + 3 < n) {
    float4 v = *(const float4*)(src + i);
    unsigned long long p =
        (unsigned long long)f32_to_bf16(v.x) |
        ((unsigned long long)f32_to_bf16(v.y) << 16) |
        ((unsigned long long)f32_to_bf16(v.z) << 32) |
        ((unsigned long long)f32_to_bf16(v.w) << 48);
    *(unsigned long long*)(dst + i) = p;
  }
}

// ---------------------------------------------------------------------------
// GEMM: C[M,N] = A[M,K] * B[K,N] + bias[N].  A,B bf16; C fp32 or bf16.
// 256 threads = 8 waves; 64x64 tile per WG; 2 16x16 output tiles per wave.
// A tile staged via async global->LDS DMA; B tile transposed via b128 loads.
// ---------------------------------------------------------------------------
template <bool OUT_BF16>
__global__ __launch_bounds__(256) void gemm_bf16(const unsigned short* __restrict__ A,
                                                 const unsigned short* __restrict__ Bm,
                                                 const float* __restrict__ bias,
                                                 void* __restrict__ Cout,
                                                 int M, int N, int K) {
  __shared__ unsigned short As[64 * 32];  // A tile, row-major (m,k)
  __shared__ unsigned short Bt[64 * 32];  // B tile transposed: (n,k)

  const int tid  = threadIdx.x;
  const int lane = tid & 31;
  const int wave = tid >> 5;
  const int tr   = wave >> 1;        // tile row 0..3
  const int tc0  = (wave & 1) * 2;   // tile cols {0,1} or {2,3}
  const int tc1  = tc0 + 1;
  const int hi   = (lane >> 4) & 1;
  const int nn   = lane & 15;
  const int row0 = blockIdx.y * 64;
  const int col0 = blockIdx.x * 64;

  v8f acc0 = {}; v8f acc1 = {};

  for (int k0 = 0; k0 < K; k0 += 32) {
    __syncthreads();  // previous iteration's readers done
    {  // A tile: 64x32 bf16 = 4KB raw copy, one async b128 per thread
      int ar = tid >> 2, ac8 = (tid & 3) * 8;
      async_copy16(&As[ar * 32 + ac8], A + (size_t)(row0 + ar) * K + k0 + ac8);
    }
    {  // B tile: 32x64 -> transpose into Bt (n,k); 8 elems per thread
      int br = tid >> 3, bc8 = (tid & 7) * 8;   // br = k row, bc8 = n offset
      const unsigned int* g =
          (const unsigned int*)(Bm + (size_t)(k0 + br) * N + col0 + bc8);
      unsigned int w[4] = {g[0], g[1], g[2], g[3]};
#pragma unroll
      for (int j = 0; j < 8; ++j)
        Bt[(bc8 + j) * 32 + br] = (unsigned short)(w[j >> 1] >> ((j & 1) * 16));
    }
    async_wait0();
    __syncthreads();

    v16bf a  = load_frag(As + tr  * 16 * 32, lane, 32);
    v16bf b0 = load_frag(Bt + tc0 * 16 * 32, lane, 32);
    v16bf b1 = load_frag(Bt + tc1 * 16 * 32, lane, 32);
    acc0 = wmma_bf16(a, b0, acc0);
    acc1 = wmma_bf16(a, b1, acc1);
  }

#pragma unroll
  for (int r = 0; r < 8; ++r) {            // C/D layout: m = r + 8*hi, n = nn
    int m  = row0 + tr * 16 + r + hi * 8;
    int c0 = col0 + tc0 * 16 + nn;
    int c1 = col0 + tc1 * 16 + nn;
    float v0 = acc0[r] + bias[c0];
    float v1 = acc1[r] + bias[c1];
    if constexpr (OUT_BF16) {
      ((unsigned short*)Cout)[(size_t)m * N + c0] = f32_to_bf16(v0);
      ((unsigned short*)Cout)[(size_t)m * N + c1] = f32_to_bf16(v1);
    } else {
      ((float*)Cout)[(size_t)m * N + c0] = v0;
      ((float*)Cout)[(size_t)m * N + c1] = v1;
    }
  }
}

// ---------------------------------------------------------------------------
// Flash-style causal attention. One WG per (b, h, 64-row query block).
// qkv: bf16, layout (b, s, t, h, c), t in {0:Q, 1:K, 2:V}.
// attn_out: bf16, layout (b, q, h, c).  1/sqrt(HS) folded into score accums.
// ---------------------------------------------------------------------------
__global__ __launch_bounds__(256) void attn_kernel(const unsigned short* __restrict__ qkv,
                                                   unsigned short* __restrict__ attn_out) {
  __shared__ unsigned short Qs[64 * 64];   // (q, c)
  __shared__ unsigned short Ks[64 * 64];   // (kpos, c)
  __shared__ unsigned short Vt[64 * 64];   // (c, kpos)  transposed
  __shared__ unsigned short Ps[64 * 64];   // (q, kpos)  softmax weights
  __shared__ float rmax_p[4][64];
  __shared__ float rsum_p[4][64];
  __shared__ float m_run[64];
  __shared__ float l_run[64];
  __shared__ float csc[64];

  const int tid  = threadIdx.x;
  const int lane = tid & 31;
  const int wave = tid >> 5;
  const int tr   = wave >> 1;
  const int tc0  = (wave & 1) * 2;
  const int tc1  = tc0 + 1;
  const int hi   = (lane >> 4) & 1;
  const int nn   = lane & 15;

  const int qb = blockIdx.x;   // query block 0..31
  const int h  = blockIdx.y;
  const int bi = blockIdx.z;
  const int q0 = qb * 64;

  {  // Q tile: 64x64 bf16 = 8KB raw async copy (layout matches global)
    const unsigned short* gq =
        qkv + (size_t)(bi * kS + q0) * kNqkv + (size_t)h * kHS;
#pragma unroll
    for (int i = 0; i < 2; ++i) {
      int idx = tid + i * 256;             // 0..511 16B chunks
      int r = idx >> 3, c8 = (idx & 7) * 8;
      async_copy16(&Qs[r * 64 + c8], gq + (size_t)r * kNqkv + c8);
    }
  }
  if (tid < 64) { m_run[tid] = -1e30f; l_run[tid] = 0.0f; }

  v8f o0 = {}; v8f o1 = {};

  for (int kb = 0; kb <= qb; ++kb) {       // causal: only kb <= qb
    __syncthreads();                       // Ks/Vt/Ps safe to overwrite
    const int k0 = kb * 64;
    const unsigned short* gk =
        qkv + (size_t)(bi * kS + k0) * kNqkv + kH * kHS + (size_t)h * kHS;
    const unsigned short* gv = gk + kH * kHS;
#pragma unroll
    for (int i = 0; i < 2; ++i) {
      int idx = tid + i * 256;
      int r = idx >> 3, c8 = (idx & 7) * 8;
      // K: raw async copy
      async_copy16(&Ks[r * 64 + c8], gk + (size_t)r * kNqkv + c8);
      // V: transpose via b128 load + b16 scatter
      const unsigned int* g = (const unsigned int*)(gv + (size_t)r * kNqkv + c8);
      unsigned int w[4] = {g[0], g[1], g[2], g[3]};
#pragma unroll
      for (int j = 0; j < 8; ++j)
        Vt[(c8 + j) * 64 + r] = (unsigned short)(w[j >> 1] >> ((j & 1) * 16));
    }
    async_wait0();                         // covers Q (first pass) + K
    __syncthreads();

    // --- scores S = Q * K^T : two 16x16 tiles per wave, HS = 2 k-chunks
    v8f s0 = {}; v8f s1 = {};
#pragma unroll
    for (int kc = 0; kc < 2; ++kc) {
      v16bf a  = load_frag(Qs + tr  * 16 * 64 + kc * 32, lane, 64);
      v16bf b0 = load_frag(Ks + tc0 * 16 * 64 + kc * 32, lane, 64);
      v16bf b1 = load_frag(Ks + tc1 * 16 * 64 + kc * 32, lane, 64);
      s0 = wmma_bf16(a, b0, s0);
      s1 = wmma_bf16(a, b1, s1);
    }

    // --- scale + causal mask + per-row max
    const bool diag = (kb == qb);
    float t0[8], t1[8];
#pragma unroll
    for (int r = 0; r < 8; ++r) {
      int m = r + hi * 8;
      float v0 = s0[r] * 0.125f;           // rsqrt(HS) folded here
      float v1 = s1[r] * 0.125f;
      if (diag && (tc0 * 16 + nn > tr * 16 + m)) v0 -= 1e9f;
      if (diag && (tc1 * 16 + nn > tr * 16 + m)) v1 -= 1e9f;
      t0[r] = v0; t1[r] = v1;
#pragma unroll
      for (int d = 8; d >= 1; d >>= 1) {
        v0 = fmaxf(v0, __shfl_xor(v0, d, 16));
        v1 = fmaxf(v1, __shfl_xor(v1, d, 16));
      }
      if (nn == 0) {
        int row = tr * 16 + m;
        rmax_p[tc0][row] = v0;
        rmax_p[tc1][row] = v1;
      }
    }
    __syncthreads();

    if (tid < 64) {                        // online-softmax row stats
      float bm = fmaxf(fmaxf(rmax_p[0][tid], rmax_p[1][tid]),
                       fmaxf(rmax_p[2][tid], rmax_p[3][tid]));
      float mo = m_run[tid];
      float mn = fmaxf(mo, bm);
      m_run[tid] = mn;
      csc[tid]   = __expf(mo - mn);
    }
    __syncthreads();

    float mn_r[8], cs_r[8];
#pragma unroll
    for (int r = 0; r < 8; ++r) {
      int row = tr * 16 + r + hi * 8;
      mn_r[r] = m_run[row];
      cs_r[r] = csc[row];
    }
#pragma unroll
    for (int r = 0; r < 8; ++r) {
      int row = tr * 16 + r + hi * 8;
      float p0 = __expf(t0[r] - mn_r[r]);
      float p1 = __expf(t1[r] - mn_r[r]);
      Ps[row * 64 + tc0 * 16 + nn] = f32_to_bf16(p0);
      Ps[row * 64 + tc1 * 16 + nn] = f32_to_bf16(p1);
      float u0 = p0, u1 = p1;
#pragma unroll
      for (int d = 8; d >= 1; d >>= 1) {
        u0 += __shfl_xor(u0, d, 16);
        u1 += __shfl_xor(u1, d, 16);
      }
      if (nn == 0) {
        rsum_p[tc0][row] = u0;
        rsum_p[tc1][row] = u1;
      }
      o0[r] *= cs_r[r];                    // rescale running output
      o1[r] *= cs_r[r];
    }
    __syncthreads();

    if (tid < 64) {
      l_run[tid] = l_run[tid] * csc[tid] +
                   rsum_p[0][tid] + rsum_p[1][tid] +
                   rsum_p[2][tid] + rsum_p[3][tid];
    }

    // --- O += P * V : key dim 64 = 2 k-chunks
#pragma unroll
    for (int kc = 0; kc < 2; ++kc) {
      v16bf a  = load_frag(Ps + tr  * 16 * 64 + kc * 32, lane, 64);
      v16bf b0 = load_frag(Vt + tc0 * 16 * 64 + kc * 32, lane, 64);
      v16bf b1 = load_frag(Vt + tc1 * 16 * 64 + kc * 32, lane, 64);
      o0 = wmma_bf16(a, b0, o0);
      o1 = wmma_bf16(a, b1, o1);
    }
  }
  __syncthreads();

  float l_r[8];
#pragma unroll
  for (int r = 0; r < 8; ++r) l_r[r] = l_run[tr * 16 + r + hi * 8];
#pragma unroll
  for (int r = 0; r < 8; ++r) {
    int q = q0 + tr * 16 + r + hi * 8;
    size_t base = ((size_t)(bi * kS + q) * kH + h) * kHS;
    attn_out[base + tc0 * 16 + nn] = f32_to_bf16(o0[r] / l_r[r]);
    attn_out[base + tc1 * 16 + nn] = f32_to_bf16(o1[r] / l_r[r]);
  }
}

// ---------------------------------------------------------------------------
extern "C" void kernel_launch(void* const* d_in, const int* in_sizes, int n_in,
                              void* d_out, int out_size, void* d_ws, size_t ws_size,
                              hipStream_t stream) {
  const float* x      = (const float*)d_in[0];
  // d_in[1] = mask (implicit via causal loop)
  const float* w_qkv  = (const float*)d_in[2];   // (D,3,H,HS) = [1024 x 3072] row-major
  const float* b_qkv  = (const float*)d_in[3];
  const float* w_proj = (const float*)d_in[4];   // (H,HS,D)   = [1024 x 1024] row-major
  const float* b_proj = (const float*)d_in[5];
  float*       out    = (float*)d_out;           // (B, S, D) fp32

  // Workspace carve-up (bf16 everywhere): ~50.3 MB total
  unsigned short* xb     = (unsigned short*)d_ws;            // 4096x1024
  unsigned short* wqkvb  = xb     + (size_t)kM * kD;         // 1024x3072
  unsigned short* wprojb = wqkvb  + (size_t)kD * kNqkv;      // 1024x1024
  unsigned short* qkvb   = wprojb + (size_t)kD * kD;         // 4096x3072
  unsigned short* attnb  = qkvb   + (size_t)kM * kNqkv;      // 4096x1024

  dim3 blk(256);

  // 0) one-time fp32 -> bf16 conversion of inputs/weights
  cvt_f32_bf16<<<(kM * kD)    / 1024, blk, 0, stream>>>(x,      xb,     kM * kD);
  cvt_f32_bf16<<<(kD * kNqkv) / 1024, blk, 0, stream>>>(w_qkv,  wqkvb,  kD * kNqkv);
  cvt_f32_bf16<<<(kD * kD)    / 1024, blk, 0, stream>>>(w_proj, wprojb, kD * kD);

  // 1) QKV projection: [4096x1024] x [1024x3072] + bias -> bf16
  gemm_bf16<true><<<dim3(kNqkv / 64, kM / 64), blk, 0, stream>>>(
      xb, wqkvb, b_qkv, qkvb, kM, kNqkv, kD);

  // 2) causal flash attention per (b, h, q-block) -> bf16
  attn_kernel<<<dim3(kS / 64, kH, kB), blk, 0, stream>>>(qkvb, attnb);

  // 3) output projection: [4096x1024] x [1024x1024] + bias -> fp32
  gemm_bf16<false><<<dim3(kD / 64, kM / 64), blk, 0, stream>>>(
      attnb, wprojb, b_proj, out, kM, kD, kD);
}
